// GemmaAttention_12275016532416
// MI455X (gfx1250) — compile-verified, matrix operand order checked by reading, not run on hardware
//
#include <hip/hip_runtime.h>
#include <hip/hip_bf16.h>
#include <stdint.h>

#define B_    2
#define S_    4096
#define DIM_  2048
#define NH_   16
#define NKV_  8
#define HD_   128
#define BK_   512
#define NB_   8
#define QKV_N 4096
#define EPS_  1e-6f
#define SCALE_ 0.08838834764831845f /* 128^-0.5 */

typedef __attribute__((ext_vector_type(16))) __bf16 v16bf;
typedef __attribute__((ext_vector_type(8)))  float  v8f;

union FragU { uint4 q[2]; v16bf v; };

__device__ __forceinline__ unsigned short f2bf(float f) {
  unsigned int u = __builtin_bit_cast(unsigned int, f);
  u += 0x7fffu + ((u >> 16) & 1u);   // round-to-nearest-even
  return (unsigned short)(u >> 16);
}

// Load a 16-element bf16 WMMA operand fragment: two 16B chunks at +0 and +16 elems.
__device__ __forceinline__ v16bf ld_frag(const unsigned short* p) {
  FragU f;
  f.q[0] = *(const uint4*)(p);
  f.q[1] = *(const uint4*)(p + 16);
  return f.v;
}

__device__ __forceinline__ v8f wmma_bf16(v16bf a, v16bf b, v8f c) {
  return __builtin_amdgcn_wmma_f32_16x16x32_bf16(false, a, false, b, (short)0, c,
                                                 false, false);
}

// Async global->LDS copy of 16 bytes (one b128 per lane), tracked by ASYNCcnt.
__device__ __forceinline__ void async_g2lds_b128(const void* gptr, void* lds_generic) {
  unsigned lds_off = (unsigned)(size_t)lds_generic;  // low 32b = LDS byte offset
  asm volatile("global_load_async_to_lds_b128 %0, %1, off"
               :: "v"(lds_off), "v"(gptr) : "memory");
}

__device__ __forceinline__ void wait_asynccnt0() {
#if defined(__has_builtin)
#if __has_builtin(__builtin_amdgcn_s_wait_asynccnt)
  __builtin_amdgcn_s_wait_asynccnt(0);
  return;
#else
  asm volatile("s_wait_asynccnt 0x0" ::: "memory");
  return;
#endif
#else
  asm volatile("s_wait_asynccnt 0x0" ::: "memory");
#endif
}

// ---------------------------------------------------------------------------
// NT GEMM: C[M,N] (f32) = A[M,K] * B[N,K]^T, bf16 WMMA w/ f32 accumulation.
// Block tile 128x128, K-step 32, 8 waves arranged 4(M) x 2(N).
// ---------------------------------------------------------------------------
template<bool A_BF16>
__global__ __launch_bounds__(256)
void gemm_nt_bf16(const void* __restrict__ Aptr, const float* __restrict__ Bptr,
                  float* __restrict__ Cptr, int M, int N, int K) {
  __shared__ unsigned short At[128 * 40];
  __shared__ unsigned short Bt[128 * 40];

  const int tid  = threadIdx.x;
  const int lane = tid & 31;
  const int wv   = tid >> 5;
  const int mw   = wv >> 1;   // 0..3  (32-row slab)
  const int nw   = wv & 1;    // 0..1  (64-col slab)
  const int ln   = lane & 15;
  const int k0   = (lane < 16) ? 0 : 8;
  const int rowAdd = (lane >= 16) ? 8 : 0;
  const int nBase = blockIdx.x * 128;
  const int mBase = blockIdx.y * 128;

  v8f acc[2][4];
#pragma unroll
  for (int mi = 0; mi < 2; ++mi)
#pragma unroll
    for (int ni = 0; ni < 4; ++ni) acc[mi][ni] = (v8f){};

  for (int kb = 0; kb < K; kb += 32) {
    __syncthreads();
    // --- stage A tile (128x32) into LDS as bf16 ---
    if (A_BF16) {
      const unsigned short* A = (const unsigned short*)Aptr;
#pragma unroll
      for (int it = 0; it < 4; ++it) {
        int e = tid + 256 * it;        // 1024 uint2 units (4 bf16 each)
        int row = e >> 3, u = e & 7;
        uint2 d = *(const uint2*)(A + (size_t)(mBase + row) * K + kb + u * 4);
        *(uint2*)(&At[row * 40 + u * 4]) = d;
      }
    } else {
      const float* A = (const float*)Aptr;
#pragma unroll
      for (int it = 0; it < 4; ++it) {
        int e = tid + 256 * it;        // 1024 float4 units
        int row = e >> 3, u = e & 7;
        float4 d = *(const float4*)(A + (size_t)(mBase + row) * K + kb + u * 4);
        uint2 h;
        h.x = (unsigned)f2bf(d.x) | ((unsigned)f2bf(d.y) << 16);
        h.y = (unsigned)f2bf(d.z) | ((unsigned)f2bf(d.w) << 16);
        *(uint2*)(&At[row * 40 + u * 4]) = h;
      }
    }
    // --- stage B tile (128x32) into LDS as bf16 ---
#pragma unroll
    for (int it = 0; it < 4; ++it) {
      int e = tid + 256 * it;
      int row = e >> 3, u = e & 7;
      float4 d = *(const float4*)(Bptr + (size_t)(nBase + row) * K + kb + u * 4);
      uint2 h;
      h.x = (unsigned)f2bf(d.x) | ((unsigned)f2bf(d.y) << 16);
      h.y = (unsigned)f2bf(d.z) | ((unsigned)f2bf(d.w) << 16);
      *(uint2*)(&Bt[row * 40 + u * 4]) = h;
    }
    __syncthreads();

    // prefetch next K-slab (global_prefetch_b8)
    if (kb + 32 < K) {
      __builtin_prefetch((const char*)Bptr + ((size_t)(nBase + (tid >> 1)) * K + kb + 32) * 4, 0, 1);
      if (!A_BF16)
        __builtin_prefetch((const char*)Aptr + ((size_t)(mBase + (tid >> 1)) * K + kb + 32) * 4, 0, 1);
    }

    v16bf af[2];
#pragma unroll
    for (int mi = 0; mi < 2; ++mi)
      af[mi] = ld_frag(&At[(mw * 32 + mi * 16 + ln) * 40 + k0]);
#pragma unroll
    for (int ni = 0; ni < 4; ++ni) {
      v16bf bf = ld_frag(&Bt[(nw * 64 + ni * 16 + ln) * 40 + k0]);
#pragma unroll
      for (int mi = 0; mi < 2; ++mi)
        acc[mi][ni] = wmma_bf16(af[mi], bf, acc[mi][ni]);
    }
  }

  // epilogue: C/D layout -> global f32
#pragma unroll
  for (int mi = 0; mi < 2; ++mi)
#pragma unroll
    for (int ni = 0; ni < 4; ++ni)
#pragma unroll
      for (int r = 0; r < 8; ++r) {
        int row = mBase + mw * 32 + mi * 16 + r + rowAdd;
        int col = nBase + nw * 64 + ni * 16 + ln;
        Cptr[(size_t)row * N + col] = acc[mi][ni][r];
      }
}

// ---------------------------------------------------------------------------
// Per-head RMSNorm + RoPE + bf16 layout transform. One wave per (b,s,head).
// heads 0..15 -> Q (norm+rope), 16..23 -> K (norm+rope), 24..31 -> V (copy).
// ---------------------------------------------------------------------------
__global__ __launch_bounds__(256)
void norm_rope(const float* __restrict__ qkv, const float* __restrict__ qnw,
               const float* __restrict__ knw, const float* __restrict__ fcos,
               const float* __restrict__ fsin, unsigned short* __restrict__ Qw,
               unsigned short* __restrict__ Kw, unsigned short* __restrict__ Vw) {
  const int tid = threadIdx.x, lane = tid & 31, wv = tid >> 5;
  const long unit = (long)blockIdx.x * 8 + wv;   // (b*S + s)*32 + head
  const int head = (int)(unit & 31);
  const int s = (int)((unit >> 5) & (S_ - 1));
  const int b = (int)(unit >> 17);
  const int i0 = lane * 4;

  const float* row = qkv + ((size_t)(b * S_ + s)) * QKV_N + head * HD_;
  float4 x = *(const float4*)(row + i0);

  if (head < 24) {
    float ss = x.x * x.x + x.y * x.y + x.z * x.z + x.w * x.w;
#pragma unroll
    for (int off = 16; off >= 1; off >>= 1) ss += __shfl_xor(ss, off, 32);
    const float inv = rsqrtf(ss * (1.0f / 128.0f) + EPS_);
    const float* nw = (head < 16) ? qnw : knw;
    float y[4], xin[4] = {x.x, x.y, x.z, x.w};
    unsigned short oh[4];
#pragma unroll
    for (int i = 0; i < 4; ++i) {
      y[i] = xin[i] * inv * (1.0f + nw[i0 + i]);
    }
#pragma unroll
    for (int i = 0; i < 4; ++i) {
      float part = __shfl_xor(y[i], 16, 32);          // pair element at +-64
      int fi = (lane & 15) * 4 + i;                   // rotary frequency index
      float c = fcos[s * 64 + fi], sn = fsin[s * 64 + fi];
      float o = (lane < 16) ? (y[i] * c - part * sn)  // real half
                            : (part * sn + y[i] * c); // imag half
      oh[i] = f2bf(o);
    }
    uint2 pk;
    pk.x = (unsigned)oh[0] | ((unsigned)oh[1] << 16);
    pk.y = (unsigned)oh[2] | ((unsigned)oh[3] << 16);
    if (head < 16)
      *(uint2*)(Qw + ((size_t)(b * NH_ + head) * S_ + s) * HD_ + i0) = pk;
    else
      *(uint2*)(Kw + ((size_t)(b * NKV_ + head - 16) * S_ + s) * HD_ + i0) = pk;
  } else {
    uint2 pk;
    pk.x = (unsigned)f2bf(x.x) | ((unsigned)f2bf(x.y) << 16);
    pk.y = (unsigned)f2bf(x.z) | ((unsigned)f2bf(x.w) << 16);
    *(uint2*)(Vw + ((size_t)(b * NKV_ + head - 24) * S_ + s) * HD_ + i0) = pk;
  }
}

// ---------------------------------------------------------------------------
// Block-sliding-window attention, flash-style online softmax.
// Grid: (NB*4, NH, B); 256 threads; wave w owns 16 q rows of a 128-row tile.
// K tile is staged with async global->LDS b128 copies (ASYNCcnt path);
// V tile is transposed through VGPRs; P round-trips a per-wave LDS tile.
// ---------------------------------------------------------------------------
__global__ __launch_bounds__(256)
void attn_win(const unsigned short* __restrict__ Qw,
              const unsigned short* __restrict__ Kw,
              const unsigned short* __restrict__ Vw,
              unsigned short* __restrict__ Ow) {
  __shared__ unsigned short Kt[32 * 136];   // [key][hd]   (row-major)
  __shared__ unsigned short Vt[128 * 40];   // [hd][key]   (transposed)
  __shared__ unsigned short Pt[8 * 16 * 40];// per-wave P tile [16][32]

  const int tid = threadIdx.x, lane = tid & 31, wv = tid >> 5;
  const int ln = lane & 15;
  const int k0 = (lane < 16) ? 0 : 8;
  const int rowAdd = (lane >= 16) ? 8 : 0;
  const int qt = blockIdx.x & 3, n = blockIdx.x >> 2;
  const int h = blockIdx.y, b = blockIdx.z;
  const int kvh = h >> 1;                   // G = 2
  const int sBase = n * BK_ + qt * 128;

  // Q fragments for this wave's 16 rows (direct global loads in A-frag layout)
  const unsigned short* qrow =
      Qw + ((size_t)(b * NH_ + h) * S_ + sBase + wv * 16 + ln) * HD_;
  v16bf qf[4];
#pragma unroll
  for (int kc = 0; kc < 4; ++kc) qf[kc] = ld_frag(qrow + kc * 32 + k0);

  v8f o[8];
#pragma unroll
  for (int f = 0; f < 8; ++f) o[f] = (v8f){};
  float m[8], l[8];
#pragma unroll
  for (int r = 0; r < 8; ++r) { m[r] = -1e30f; l[r] = 0.0f; }

  const int jstart = (n == 0) ? 512 : qt * 128;
  const int jend = 640 + qt * 128;
  const int lqBase = qt * 128 + wv * 16 + rowAdd;

  for (int j = jstart; j < jend; j += 32) {
    const int skBase = (n - 1) * BK_ + j;   // >= 0 (n==0 => j>=512)
    __syncthreads();
    // --- K chunk (32x128 bf16): async global->LDS, 2 x b128 per lane ---
    const unsigned short* Ksrc = Kw + ((size_t)(b * NKV_ + kvh) * S_ + skBase) * HD_;
#pragma unroll
    for (int it = 0; it < 2; ++it) {
      int e = tid + 256 * it;               // 512 b128 units (8 bf16 each)
      int key = e >> 4, u = e & 15;
      async_g2lds_b128(Ksrc + (size_t)key * HD_ + u * 8, &Kt[key * 136 + u * 8]);
    }
    // --- V chunk, transposed into LDS through VGPRs ---
    const unsigned short* Vsrc = Vw + ((size_t)(b * NKV_ + kvh) * S_ + skBase) * HD_;
#pragma unroll
    for (int it = 0; it < 4; ++it) {
      int e = tid + 256 * it;
      int key = e >> 5, u = e & 31;
      uint2 d = *(const uint2*)(Vsrc + (size_t)key * HD_ + u * 4);
      Vt[(u * 4 + 0) * 40 + key] = (unsigned short)(d.x & 0xffffu);
      Vt[(u * 4 + 1) * 40 + key] = (unsigned short)(d.x >> 16);
      Vt[(u * 4 + 2) * 40 + key] = (unsigned short)(d.y & 0xffffu);
      Vt[(u * 4 + 3) * 40 + key] = (unsigned short)(d.y >> 16);
    }
    wait_asynccnt0();          // drain this wave's async copies before barrier
    __syncthreads();

    // S = Q * K^T  (16 q-rows x 32 keys), 4 K-steps over HD=128
    v8f s[2];
    s[0] = (v8f){}; s[1] = (v8f){};
#pragma unroll
    for (int kc = 0; kc < 4; ++kc) {
#pragma unroll
      for (int ni = 0; ni < 2; ++ni) {
        v16bf kf = ld_frag(&Kt[(ni * 16 + ln) * 136 + kc * 32 + k0]);
        s[ni] = wmma_bf16(qf[kc], kf, s[ni]);
      }
    }
    // scale + mask
#pragma unroll
    for (int ni = 0; ni < 2; ++ni) {
      int jcol = j + ni * 16 + ln;
#pragma unroll
      for (int r = 0; r < 8; ++r) {
        int lq = lqBase + r;
        bool ok = (jcol >= lq) && (jcol <= 512 + lq);
        s[ni][r] = ok ? (s[ni][r] * SCALE_) : -1e30f;
      }
    }
    // online softmax per row (16-lane half reductions)
#pragma unroll
    for (int r = 0; r < 8; ++r) {
      float cm = fmaxf(s[0][r], s[1][r]);
#pragma unroll
      for (int off = 8; off >= 1; off >>= 1) cm = fmaxf(cm, __shfl_xor(cm, off, 32));
      float nm = fmaxf(m[r], cm);
      float sf = __expf(m[r] - nm);
      m[r] = nm;
      float p0 = (s[0][r] <= -1e29f) ? 0.0f : __expf(s[0][r] - nm);
      float p1 = (s[1][r] <= -1e29f) ? 0.0f : __expf(s[1][r] - nm);
      float rs = p0 + p1;
#pragma unroll
      for (int off = 8; off >= 1; off >>= 1) rs += __shfl_xor(rs, off, 32);
      l[r] = l[r] * sf + rs;
#pragma unroll
      for (int f = 0; f < 8; ++f) o[f][r] *= sf;
      // P (C-layout) -> per-wave LDS tile (A-layout source)
      Pt[(wv * 16 + r + rowAdd) * 40 + ln] = f2bf(p0);
      Pt[(wv * 16 + r + rowAdd) * 40 + 16 + ln] = f2bf(p1);
    }
    // O += P * V   (K = 32 keys)
    v16bf pf = ld_frag(&Pt[(wv * 16 + ln) * 40 + k0]);
#pragma unroll
    for (int f = 0; f < 8; ++f) {
      v16bf vf = ld_frag(&Vt[(f * 16 + ln) * 40 + k0]);
      o[f] = wmma_bf16(pf, vf, o[f]);
    }
  }

  // normalize + store bf16 into attention-out [B, S, NH*HD]
#pragma unroll
  for (int f = 0; f < 8; ++f)
#pragma unroll
    for (int r = 0; r < 8; ++r) {
      int srow = sBase + wv * 16 + r + rowAdd;
      int col = h * HD_ + f * 16 + ln;
      Ow[(size_t)(b * S_ + srow) * (NH_ * HD_) + col] = f2bf(o[f][r] / l[r]);
    }
}

// ---------------------------------------------------------------------------
extern "C" void kernel_launch(void* const* d_in, const int* in_sizes, int n_in,
                              void* d_out, int out_size, void* d_ws, size_t ws_size,
                              hipStream_t stream) {
  (void)in_sizes; (void)n_in; (void)out_size; (void)ws_size;
  const float* hidden = (const float*)d_in[0];
  const float* w_qkv  = (const float*)d_in[1];
  const float* w_o    = (const float*)d_in[2];
  const float* qnw    = (const float*)d_in[3];
  const float* knw    = (const float*)d_in[4];
  const float* fcos   = (const float*)d_in[5];
  const float* fsin   = (const float*)d_in[6];
  float* out = (float*)d_out;

  char* ws = (char*)d_ws;
  float*          qkv = (float*)(ws);                       // 128 MiB f32
  unsigned short* Qw  = (unsigned short*)(ws + 134217728);  //  32 MiB bf16
  unsigned short* Kw  = (unsigned short*)(ws + 167772160);  //  16 MiB bf16
  unsigned short* Vw  = (unsigned short*)(ws + 184549376);  //  16 MiB bf16
  unsigned short* Aw  = (unsigned short*)(ws + 201326592);  //  32 MiB bf16

  dim3 blk(256);
  // 1) QKV projection: [8192,2048] x [4096,2048]^T -> [8192,4096]
  gemm_nt_bf16<false><<<dim3(QKV_N / 128, (B_ * S_) / 128), blk, 0, stream>>>(
      hidden, w_qkv, qkv, B_ * S_, QKV_N, DIM_);
  // 2) RMSNorm + RoPE + bf16 head-major layout
  norm_rope<<<dim3((B_ * S_ * 32) / 8), blk, 0, stream>>>(
      qkv, qnw, knw, fcos, fsin, Qw, Kw, Vw);
  // 3) windowed attention
  attn_win<<<dim3(NB_ * 4, NH_, B_), blk, 0, stream>>>(Qw, Kw, Vw, Aw);
  // 4) output projection: [8192,2048] x [2048,2048]^T -> [8192,2048]
  gemm_nt_bf16<true><<<dim3(DIM_ / 128, (B_ * S_) / 128), blk, 0, stream>>>(
      Aw, w_o, out, B_ * S_, DIM_, NH_ * HD_);
}
